// DeXPaReClassifier_83030307766372
// MI455X (gfx1250) — compile-verified, matrix operand order checked by reading, not run on hardware
//
#include <hip/hip_runtime.h>
#include <math.h>

// ---------------------------------------------------------------------------
// Types for WMMA (gfx1250, wave32)
// ---------------------------------------------------------------------------
typedef __attribute__((ext_vector_type(16))) __bf16 v16bf;
typedef __attribute__((ext_vector_type(8)))  __bf16 v8bf;
typedef __attribute__((ext_vector_type(8)))  float  v8f;

__device__ inline v8f vzero8() {
    v8f z;
#pragma unroll
    for (int i = 0; i < 8; ++i) z[i] = 0.0f;
    return z;
}

// ---------------------------------------------------------------------------
// Block-wide reductions (blockDim.x == 256)
// ---------------------------------------------------------------------------
__device__ inline float block_sum(float v, float* red) {
    int t = threadIdx.x;
    red[t] = v; __syncthreads();
    for (int s = 128; s > 0; s >>= 1) {
        if (t < s) red[t] += red[t + s];
        __syncthreads();
    }
    float r = red[0]; __syncthreads();
    return r;
}

__device__ inline float block_max(float v, float* red) {
    int t = threadIdx.x;
    red[t] = v; __syncthreads();
    for (int s = 128; s > 0; s >>= 1) {
        if (t < s) red[t] = fmaxf(red[t], red[t + s]);
        __syncthreads();
    }
    float r = red[0]; __syncthreads();
    return r;
}

// ---------------------------------------------------------------------------
// Generic WMMA GEMM:  C[M x N] = A[M x K] * B[N x K]^T  (+ epilogue per MODE)
//   MODE 0: tex_fc  = tex_n_bf16 * fc_w_bf16^T + fc_b      -> fp32 (M=8000 guard)
//   MODE 1: cp      = (img_f * invnorm) * tex_a^T          -> bf16, col base z*128
//   MODE 2: h       = elu(cp * w1'^T + b1')                -> bf16
//   MODE 3: out     = h * w2''^T + b2''                    -> fp32 (col<1000 guard)
// Block: 256 threads = 8 waves (4x2), block tile 128x128, K-step 32.
// Each wave computes 32x64 (2x4 WMMA subtiles): 8 WMMAs / 12 ds_load_b128.
// ---------------------------------------------------------------------------
template <int MODE>
__global__ void __launch_bounds__(256)
gemm_wmma_k(const float* __restrict__ Af32,
            const __bf16* __restrict__ Abf,
            const __bf16* __restrict__ Bbf,
            const float* __restrict__ bias,
            const float* __restrict__ scl,
            float* __restrict__ Cf,
            __bf16* __restrict__ Cb,
            int M, int K, int lda)
{
    constexpr int MT = 128, NT = 128, KT = 32, LW = 40;  // LDS row stride (bf16)
    __shared__ __align__(16) __bf16 sA[MT * LW];
    __shared__ __align__(16) __bf16 sB[NT * LW];

    const int tid = threadIdx.x;
    const int m0  = blockIdx.x * MT;
    const int nB0 = blockIdx.y * NT;        // B-row base
    int colBase = nB0;                      // output column base
    int aoff    = 0;
    if constexpr (MODE == 1) {
        aoff    = blockIdx.z * 512;         // x-slice of img_f
        colBase = blockIdx.z * 128 + nB0;   // cp column layout: x*128 + p*16 + d
    }

    // staging decomposition: thread -> (row, 16-elem half) for both A and B
    const int sr = tid >> 1;                // row 0..127
    const int sh = (tid & 1) * 16;          // half (elements)

    // wave / lane decomposition
    const int wave = tid >> 5, lane = tid & 31;
    const int wm = wave >> 1, wn = wave & 1;  // 4 x 2 wave grid
    const int lr  = lane & 15;
    const int kof = (lane & 16) ? 8 : 0;    // K-half select (also row+8 in C layout)

    v8f acc[2][4];
#pragma unroll
    for (int i = 0; i < 2; ++i)
#pragma unroll
        for (int j = 0; j < 4; ++j) acc[i][j] = vzero8();

    float ascale = 0.0f;
    if constexpr (MODE == 1) ascale = scl[(size_t)(m0 + sr) * 8 + blockIdx.z];

    for (int k0 = 0; k0 < K; k0 += KT) {
        // ---- stage A tile (128 x 32 bf16) ----
        if constexpr (MODE == 1) {
            const float* ap = Af32 + (size_t)(m0 + sr) * lda + aoff + k0 + sh;
            __bf16* dst = sA + sr * LW + sh;
#pragma unroll
            for (int i = 0; i < 4; ++i) {
                float4 f = *(const float4*)(ap + i * 4);
                dst[i * 4 + 0] = (__bf16)(f.x * ascale);
                dst[i * 4 + 1] = (__bf16)(f.y * ascale);
                dst[i * 4 + 2] = (__bf16)(f.z * ascale);
                dst[i * 4 + 3] = (__bf16)(f.w * ascale);
            }
            if (k0 + KT < K) __builtin_prefetch(ap + KT, 0, 1);
        } else {
            const int row = m0 + sr;
            v8bf z0, z1;
            if (MODE == 0 && row >= M) {
#pragma unroll
                for (int i = 0; i < 8; ++i) { z0[i] = (__bf16)0.0f; z1[i] = (__bf16)0.0f; }
            } else {
                const __bf16* ap = Abf + (size_t)row * lda + k0 + sh;
                z0 = *(const v8bf*)ap;
                z1 = *(const v8bf*)(ap + 8);
                if (k0 + KT < K) __builtin_prefetch(ap + KT, 0, 1);
            }
            *(v8bf*)(sA + sr * LW + sh)     = z0;
            *(v8bf*)(sA + sr * LW + sh + 8) = z1;
        }
        // ---- stage B tile (128 x 32 bf16) ----
        {
            const __bf16* bp = Bbf + (size_t)(nB0 + sr) * K + k0 + sh;
            *(v8bf*)(sB + sr * LW + sh)     = *(const v8bf*)bp;
            *(v8bf*)(sB + sr * LW + sh + 8) = *(const v8bf*)(bp + 8);
            if (k0 + KT < K) __builtin_prefetch(bp + KT, 0, 1);
        }
        __syncthreads();

        // ---- fragments + 8 WMMAs per wave ----
        v16bf aF[2], bF[4];
#pragma unroll
        for (int i = 0; i < 2; ++i) {
            const __bf16* p = sA + (wm * 32 + i * 16 + lr) * LW + kof;
            v8bf lo = *(const v8bf*)p;
            v8bf hi = *(const v8bf*)(p + 16);
            aF[i] = __builtin_shufflevector(lo, hi, 0, 1, 2, 3, 4, 5, 6, 7,
                                            8, 9, 10, 11, 12, 13, 14, 15);
        }
#pragma unroll
        for (int j = 0; j < 4; ++j) {
            const __bf16* p = sB + (wn * 64 + j * 16 + lr) * LW + kof;
            v8bf lo = *(const v8bf*)p;
            v8bf hi = *(const v8bf*)(p + 16);
            bF[j] = __builtin_shufflevector(lo, hi, 0, 1, 2, 3, 4, 5, 6, 7,
                                            8, 9, 10, 11, 12, 13, 14, 15);
        }
#pragma unroll
        for (int i = 0; i < 2; ++i)
#pragma unroll
            for (int j = 0; j < 4; ++j)
                acc[i][j] = __builtin_amdgcn_wmma_f32_16x16x32_bf16(
                    false, aF[i], false, bF[j], (short)0, acc[i][j], false, false);
        __syncthreads();
    }

    // ---- epilogue ----
    constexpr int ldc = (MODE == 0) ? 512 : (MODE == 1) ? 1024 : (MODE == 2) ? 2048 : 1000;
#pragma unroll
    for (int j = 0; j < 4; ++j) {
        const int col = colBase + wn * 64 + j * 16 + lr;
        float bb = 0.0f;
        if constexpr (MODE == 0 || MODE == 2 || MODE == 3) bb = bias[col];
#pragma unroll
        for (int i = 0; i < 2; ++i) {
            const int rowb = m0 + wm * 32 + i * 16 + kof;  // lanes>=16 -> +8 rows
#pragma unroll
            for (int r = 0; r < 8; ++r) {
                const int row = rowb + r;
                const float v = acc[i][j][r];
                if constexpr (MODE == 0) {
                    if (row < M) Cf[(size_t)row * ldc + col] = v + bb;
                } else if constexpr (MODE == 1) {
                    Cb[(size_t)row * ldc + col] = (__bf16)v;
                } else if constexpr (MODE == 2) {
                    float t = v + bb;
                    float e = (t > 0.0f) ? t : (__expf(t) - 1.0f);   // ELU
                    Cb[(size_t)row * ldc + col] = (__bf16)e;
                } else {
                    if (col < 1000) Cf[(size_t)row * ldc + col] = v + bb;
                }
            }
        }
    }
}

// ---------------------------------------------------------------------------
// tex_f rows: L2-normalize * 64 -> bf16. One block per row (512 elems).
// ---------------------------------------------------------------------------
__global__ void __launch_bounds__(256)
k_tex_norm(const float* __restrict__ tex, __bf16* __restrict__ out)
{
    __shared__ float red[256];
    const int row = blockIdx.x, t = threadIdx.x;
    const float x0 = tex[(size_t)row * 512 + t];
    const float x1 = tex[(size_t)row * 512 + 256 + t];
    const float ss = block_sum(x0 * x0 + x1 * x1, red);
    const float sc = 64.0f / (sqrtf(ss) + 1e-6f);
    out[(size_t)row * 512 + t]       = (__bf16)(x0 * sc);
    out[(size_t)row * 512 + 256 + t] = (__bf16)(x1 * sc);
}

// fp32 -> bf16 convert
__global__ void __launch_bounds__(256)
k_cvt_bf16(const float* __restrict__ in, __bf16* __restrict__ out, int n)
{
    int i = blockIdx.x * 256 + threadIdx.x;
    if (i < n) out[i] = (__bf16)in[i];
}

// img_f row inv-norms (*64): one wave32 per (b,x) row of 512.
__global__ void __launch_bounds__(256)
k_img_norm(const float* __restrict__ img, float* __restrict__ invn)
{
    const int row  = blockIdx.x * 8 + (threadIdx.x >> 5);
    const int lane = threadIdx.x & 31;
    float ss = 0.0f;
    for (int c = lane; c < 512; c += 32) {
        float x = img[(size_t)row * 512 + c];
        ss += x * x;
    }
#pragma unroll
    for (int m = 16; m > 0; m >>= 1) ss += __shfl_xor(ss, m);
    if (lane == 0) invn[row] = 64.0f / (sqrtf(ss) + 1e-6f);
}

// ---------------------------------------------------------------------------
// Prompt attention: per (p, fake) block — logits over Y=1000, softmax,
// weighted sum of tex_n -> tex_a (bf16).
// ---------------------------------------------------------------------------
__global__ void __launch_bounds__(256)
k_attention(const float* __restrict__ fake_cls,
            const float* __restrict__ texfc,
            const __bf16* __restrict__ texn,
            __bf16* __restrict__ texa)
{
    __shared__ float fake[512];
    __shared__ float lg[1000];
    __shared__ float red[256];
    const int p = blockIdx.x >> 4, fx = blockIdx.x & 15, t = threadIdx.x;

    for (int c = t; c < 512; c += 256)
        fake[c] = fake_cls[((size_t)(p * 16 + fx)) * 512 + c];
    __syncthreads();

    float lmax = -1e30f;
    for (int y = t; y < 1000; y += 256) {
        const float* tp = texfc + ((size_t)(p * 1000 + y)) * 512;
        float d = 0.0f;
        for (int c = 0; c < 512; ++c) d += fake[c] * tp[c];
        lg[y] = d;
        lmax = fmaxf(lmax, d);
    }
    const float gmax = block_max(lmax, red);

    float lsum = 0.0f;
    for (int y = t; y < 1000; y += 256) {
        float e = __expf(lg[y] - gmax);
        lg[y] = e;
        lsum += e;
    }
    const float inv = 1.0f / block_sum(lsum, red);
    for (int y = t; y < 1000; y += 256) lg[y] *= inv;
    __syncthreads();

    for (int c = t; c < 512; c += 256) {
        const __bf16* tn = texn + (size_t)p * 1000 * 512 + c;
        float a = 0.0f;
        for (int y = 0; y < 1000; ++y) a += lg[y] * (float)tn[(size_t)y * 512];
        texa[((size_t)(p * 16 + fx)) * 512 + c] = (__bf16)a;
    }
}

// ---------------------------------------------------------------------------
// Deterministic column statistics: partial sums / sums-of-squares per chunk.
// grid = (N/256, 32); rowsPerChunk rows each.
// ---------------------------------------------------------------------------
__global__ void __launch_bounds__(256)
k_colstat(const __bf16* __restrict__ X, float* __restrict__ ps,
          float* __restrict__ pq, int N, int rowsPerChunk)
{
    const int col = blockIdx.x * 256 + threadIdx.x;
    const int r0  = blockIdx.y * rowsPerChunk;
    float s = 0.0f, q = 0.0f;
    for (int r = 0; r < rowsPerChunk; ++r) {
        float v = (float)X[(size_t)(r0 + r) * N + col];
        s += v;
        q += v * v;
    }
    ps[(size_t)blockIdx.y * N + col] = s;
    pq[(size_t)blockIdx.y * N + col] = q;
}

// BN fold coefficients: s = rstd*gamma, t = beta - mean*rstd*gamma.
// permute!=0 maps cp layout j'=x*128+p*16+d back to reference j=p*128+x*16+d.
__global__ void __launch_bounds__(256)
k_bnfin(int n, const float* __restrict__ ps, const float* __restrict__ pq,
        const float* __restrict__ g, const float* __restrict__ bb,
        float* __restrict__ sc, float* __restrict__ tc, int permute)
{
    const int i = blockIdx.x * 256 + threadIdx.x;
    if (i >= n) return;
    float sm = 0.0f, sq = 0.0f;
    for (int p = 0; p < 32; ++p) { sm += ps[(size_t)p * n + i]; sq += pq[(size_t)p * n + i]; }
    const float mean = sm * (1.0f / 8192.0f);
    const float var  = sq * (1.0f / 8192.0f) - mean * mean;
    const float r    = rsqrtf(var + 1e-5f);
    int j = i;
    if (permute) {
        int x = i >> 7, rem = i & 127, pp = rem >> 4, d = rem & 15;
        j = pp * 128 + x * 16 + d;
    }
    const float gg = g[j] * r;
    sc[i] = gg;
    tc[i] = bb[j] - mean * gg;
}

// Fold BN1 into w1 (permuted columns) -> bf16, and b1' = b1 + sum_j t_j*w1[f,j].
__global__ void __launch_bounds__(256)
k_fold_w1(const float* __restrict__ w1, const float* __restrict__ b1,
          const float* __restrict__ s1, const float* __restrict__ t1,
          __bf16* __restrict__ w1p, float* __restrict__ b1p)
{
    __shared__ float red[256];
    const int f = blockIdx.x;
    float acc = 0.0f;
    for (int jp = threadIdx.x; jp < 1024; jp += 256) {
        int x = jp >> 7, rem = jp & 127, pp = rem >> 4, d = rem & 15;
        int j = pp * 128 + x * 16 + d;
        float wv = w1[(size_t)f * 1024 + j];
        w1p[(size_t)f * 1024 + jp] = (__bf16)(wv * s1[jp]);
        acc += t1[jp] * wv;
    }
    acc = block_sum(acc, red);
    if (threadIdx.x == 0) b1p[f] = b1[f] + acc;
}

// Fold BN2 into w2 -> bf16 (padded to 1024 rows), b2' likewise.
__global__ void __launch_bounds__(256)
k_fold_w2(const float* __restrict__ w2, const float* __restrict__ b2,
          const float* __restrict__ s2, const float* __restrict__ t2,
          __bf16* __restrict__ w2p, float* __restrict__ b2p)
{
    __shared__ float red[256];
    const int y = blockIdx.x;
    if (y < 1000) {
        float acc = 0.0f;
        for (int f = threadIdx.x; f < 2048; f += 256) {
            float wv = w2[(size_t)y * 2048 + f];
            w2p[(size_t)y * 2048 + f] = (__bf16)(wv * s2[f]);
            acc += t2[f] * wv;
        }
        acc = block_sum(acc, red);
        if (threadIdx.x == 0) b2p[y] = b2[y] + acc;
    } else {
        for (int f = threadIdx.x; f < 2048; f += 256)
            w2p[(size_t)y * 2048 + f] = (__bf16)0.0f;
        if (threadIdx.x == 0) b2p[y] = 0.0f;
    }
}

// ---------------------------------------------------------------------------
// Host launcher
// ---------------------------------------------------------------------------
extern "C" void kernel_launch(void* const* d_in, const int* in_sizes, int n_in,
                              void* d_out, int out_size, void* d_ws, size_t ws_size,
                              hipStream_t stream)
{
    (void)in_sizes; (void)n_in; (void)out_size; (void)ws_size;
    const float* img_f    = (const float*)d_in[0];
    const float* tex_f    = (const float*)d_in[1];
    const float* fake_cls = (const float*)d_in[2];
    const float* fc_w     = (const float*)d_in[3];
    const float* fc_b     = (const float*)d_in[4];
    const float* bn1_g    = (const float*)d_in[5];
    const float* bn1_b    = (const float*)d_in[6];
    const float* w1       = (const float*)d_in[7];
    const float* b1       = (const float*)d_in[8];
    const float* bn2_g    = (const float*)d_in[9];
    const float* bn2_b    = (const float*)d_in[10];
    const float* w2       = (const float*)d_in[11];
    const float* b2       = (const float*)d_in[12];
    float* out = (float*)d_out;

    char* ws = (char*)d_ws;
    size_t off = 0;
    auto alloc = [&](size_t bytes) -> char* {
        char* p = ws + off;
        off = (off + bytes + 255) & ~(size_t)255;
        return p;
    };

    __bf16* texn  = (__bf16*)alloc((size_t)8 * 1000 * 512 * 2);
    __bf16* fcwb  = (__bf16*)alloc((size_t)512 * 512 * 2);
    float*  texfc = (float*)alloc((size_t)8 * 1000 * 512 * 4);
    __bf16* texa  = (__bf16*)alloc((size_t)128 * 512 * 2);
    float*  invn  = (float*)alloc((size_t)65536 * 4);
    __bf16* cpb   = (__bf16*)alloc((size_t)8192 * 1024 * 2);
    float*  ps1   = (float*)alloc((size_t)32 * 1024 * 4);
    float*  pq1   = (float*)alloc((size_t)32 * 1024 * 4);
    float*  s1    = (float*)alloc(1024 * 4);
    float*  t1    = (float*)alloc(1024 * 4);
    __bf16* w1p   = (__bf16*)alloc((size_t)2048 * 1024 * 2);
    float*  b1p   = (float*)alloc(2048 * 4);
    __bf16* hb    = (__bf16*)alloc((size_t)8192 * 2048 * 2);
    float*  ps2   = (float*)alloc((size_t)32 * 2048 * 4);
    float*  pq2   = (float*)alloc((size_t)32 * 2048 * 4);
    float*  s2    = (float*)alloc(2048 * 4);
    float*  t2    = (float*)alloc(2048 * 4);
    __bf16* w2p   = (__bf16*)alloc((size_t)1024 * 2048 * 2);
    float*  b2p   = (float*)alloc(1024 * 4);

    // 1) tex normalization + fc_w convert
    k_tex_norm<<<8000, 256, 0, stream>>>(tex_f, texn);
    k_cvt_bf16<<<(512 * 512 + 255) / 256, 256, 0, stream>>>(fc_w, fcwb, 512 * 512);

    // 2) tex_fc = tex_n @ fc_w^T + fc_b   (M=8000, N=512, K=512)
    gemm_wmma_k<0><<<dim3(63, 4, 1), 256, 0, stream>>>(
        nullptr, texn, fcwb, fc_b, nullptr, texfc, nullptr, 8000, 512, 512);

    // 3) attention -> tex_a (128 x 512 bf16)
    k_attention<<<128, 256, 0, stream>>>(fake_cls, texfc, texn, texa);

    // 4) img row inv-norms
    k_img_norm<<<8192, 256, 0, stream>>>(img_f, invn);

    // 5) cp = img_n @ tex_a^T  (per x-slice; M=8192, N=128, K=512)
    gemm_wmma_k<1><<<dim3(64, 1, 8), 256, 0, stream>>>(
        img_f, nullptr, texa, nullptr, invn, nullptr, cpb, 8192, 512, 4096);

    // 6) BN1 stats + fold into w1
    k_colstat<<<dim3(4, 32), 256, 0, stream>>>(cpb, ps1, pq1, 1024, 256);
    k_bnfin<<<4, 256, 0, stream>>>(1024, ps1, pq1, bn1_g, bn1_b, s1, t1, 1);
    k_fold_w1<<<2048, 256, 0, stream>>>(w1, b1, s1, t1, w1p, b1p);

    // 7) h = elu(cp @ w1'^T + b1')  (M=8192, N=2048, K=1024)
    gemm_wmma_k<2><<<dim3(64, 16, 1), 256, 0, stream>>>(
        nullptr, cpb, w1p, b1p, nullptr, nullptr, hb, 8192, 1024, 1024);

    // 8) BN2 stats + fold into w2
    k_colstat<<<dim3(8, 32), 256, 0, stream>>>(hb, ps2, pq2, 2048, 256);
    k_bnfin<<<8, 256, 0, stream>>>(2048, ps2, pq2, bn2_g, bn2_b, s2, t2, 0);
    k_fold_w2<<<1024, 256, 0, stream>>>(w2, b2, s2, t2, w2p, b2p);

    // 9) out = h @ w2''^T + b2''  (M=8192, N=1024 padded, K=2048)
    gemm_wmma_k<3><<<dim3(64, 8, 1), 256, 0, stream>>>(
        nullptr, hb, w2p, b2p, nullptr, out, nullptr, 8192, 2048, 2048);
}
